// CGNet_SSM_25984552140982
// MI455X (gfx1250) — compile-verified, matrix operand order checked by reading, not run on hardware
//
#include <hip/hip_runtime.h>
#include <hip/hip_bf16.h>
#include <math.h>

typedef __attribute__((ext_vector_type(16))) _Float16 v16h;
typedef __attribute__((ext_vector_type(8)))  _Float16 v8h;
typedef __attribute__((ext_vector_type(8)))  float    v8f;
typedef __attribute__((ext_vector_type(4)))  int      v4i;

// CDNA5 async Global->LDS path (gfx1250): guarded so either toolchain compiles.
#if defined(__has_builtin)
#if __has_builtin(__builtin_amdgcn_global_load_async_to_lds_b128) && \
    __has_builtin(__builtin_amdgcn_s_wait_asynccnt)
#define CGNET_HAVE_ASYNC_LDS 1
#endif
#endif

// ============================================================================
// Weight repack: f32 (O,I,T) -> f16 in the exact 16-bit A-matrix WMMA layout.
// Block layout: for (oc_tile ot, tap t, kchunk kc): 32 lanes x 16 halves.
// half j of lane l maps to k = ((p<4)?2p:16+2(p-4)) + (l&16?8:0) + (j&1), m=l&15
// ============================================================================
__global__ void repack_w_kernel(const float* __restrict__ w, _Float16* __restrict__ wp,
                                int O, int I, int T, long total) {
  long i = (long)blockIdx.x * blockDim.x + threadIdx.x;
  long stride = (long)gridDim.x * blockDim.x;
  int KC = (I + 31) >> 5;
  for (; i < total; i += stride) {
    int j = (int)(i & 15);
    int lanex = (int)((i >> 4) & 31);
    long blk = i >> 9;
    int kc = (int)(blk % KC);
    long b2 = blk / KC;
    int t = (int)(b2 % T);
    int ot = (int)(b2 / T);
    int p = j >> 1;
    int k = ((p < 4) ? (p << 1) : (16 + ((p - 4) << 1))) + ((lanex & 16) ? 8 : 0) + (j & 1);
    int oc = (ot << 4) + (lanex & 15);
    int ci = (kc << 5) + k;
    float val = 0.f;
    if (oc < O && ci < I) val = w[((long)oc * I + ci) * T + t];
    wp[i] = (_Float16)val;
  }
}

// ============================================================================
// Fused conv (K=1 or K=3, pad=K/2) + BN(eval) or bias + optional ReLU.
// Implicit GEMM, templated on K so the tap loop fully unrolls:
//   per wave: M=32 out-channels (two 16x16 tiles sharing one B fragment)
//             x N=16 pixels (one row segment), K-loop over cin chunks of 32.
// LDS tile is f16 [row][col][ci32] so a B fragment is two ds_load_b128.
// If the input has an f16 NHWC shadow (in_h, Cpad=KC*32), staging is a pure
// 16B-segment gather issued as async global->LDS copies (ASYNCcnt); halo
// segments are zero-filled with ds stores. Otherwise (first conv) stage from
// f32 NCHW with convert-in-flight.
// Epilogue writes f32 NCHW and, if out_h != null (OC % 32 == 0), the f16 NHWC
// shadow as one 16B store per tile per lane (D rows = consecutive channels).
// ============================================================================
template <int K>
__global__ __launch_bounds__(256)
void conv_wmma_kernel(const float* __restrict__ in, const _Float16* __restrict__ in_h,
                      const _Float16* __restrict__ wpack,
                      const float* __restrict__ bn_g, const float* __restrict__ bn_b,
                      const float* __restrict__ bn_m, const float* __restrict__ bn_v,
                      const float* __restrict__ bias,
                      float* __restrict__ out, _Float16* __restrict__ out_h,
                      int N, int C, int H, int W, int OC, int relu) {
  __shared__ _Float16 smem[10 * 18 * 32];  // [row][col][ci] f16, 11.25 KB
  const int OCT  = (OC + 15) >> 4;
  const int OCPT = (OCT + 1) >> 1;
  const int n    = blockIdx.z / OCPT;
  const int ot0  = (blockIdx.z % OCPT) << 1;
  const bool has2 = (ot0 + 1) < OCT;  // wave-uniform: EXEC stays all-ones
  const int x0   = blockIdx.x << 4;
  const int ybase = blockIdx.y << 3;
  const int tid  = threadIdx.x;
  const int lane = tid & 31;
  const int wave = tid >> 5;
  const int y    = ybase + wave;
  const int KC   = (C + 31) >> 5;
  const int Cpad = KC << 5;
  const int T    = K * K;
  const int pad  = (K == 3) ? 1 : 0;
  const int npix = lane & 15;
  const int khalf = (lane & 16) ? 8 : 0;
  const long HWl = (long)H * W;
  const float* inN = in + (long)n * C * HWl;
  const _Float16* inhN = in_h ? (in_h + (long)n * HWl * Cpad) : (const _Float16*)0;

  v8f acc0 = {0.f, 0.f, 0.f, 0.f, 0.f, 0.f, 0.f, 0.f};
  v8f acc1 = {0.f, 0.f, 0.f, 0.f, 0.f, 0.f, 0.f, 0.f};

  for (int kc = 0; kc < KC; ++kc) {
    __syncthreads();
    if (inhN) {
      // 180 segments of 64B (one per (row,col)), 4 x 16B quarters each
      for (int s4 = tid; s4 < 180 * 4; s4 += 256) {
        int s = s4 >> 2, q = s4 & 3;
        int r = s / 18, cc = s % 18;
        int gy = ybase + r - 1;
        int gx = x0 + cc - 1;
        _Float16* ldst = &smem[((r * 18 + cc) << 5) + (q << 3)];
        if (gy >= 0 && gy < H && gx >= 0 && gx < W) {
          const _Float16* gsrc = inhN + ((long)gy * W + gx) * Cpad + (kc << 5) + (q << 3);
#ifdef CGNET_HAVE_ASYNC_LDS
          __builtin_amdgcn_global_load_async_to_lds_b128(
              (__attribute__((address_space(1))) v4i*)gsrc,
              (__attribute__((address_space(3))) v4i*)ldst, 0, 0);
#else
          *(v8h*)ldst = *(const v8h*)gsrc;
#endif
        } else {
          v8h z = {};
          *(v8h*)ldst = z;  // zero halo
        }
      }
#ifdef CGNET_HAVE_ASYNC_LDS
      __builtin_amdgcn_s_wait_asynccnt(0);
#endif
    } else {
      // f32 NCHW fallback (first conv): convert in flight
      for (int idx = tid; idx < 32 * 10 * 18; idx += 256) {
        int ci = idx / 180;
        int rem = idx % 180;
        int r = rem / 18;
        int cc = rem % 18;
        int gci = (kc << 5) + ci;
        int gy = ybase + r - 1;
        int gx = x0 + cc - 1;
        float val = 0.f;
        if (gci < C && gy >= 0 && gy < H && gx >= 0 && gx < W)
          val = inN[(long)gci * HWl + (long)gy * W + gx];
        smem[(r * 18 + cc) * 32 + ci] = (_Float16)val;
      }
    }
    if (kc + 1 < KC && inhN)
      __builtin_prefetch(inhN + ((long)ybase * W + x0) * Cpad + ((kc + 1) << 5), 0, 1);
    __syncthreads();

#pragma unroll
    for (int t = 0; t < T; ++t) {
      const int dy = t / K - pad;
      const int dx = t - (t / K) * K - pad;
      const int rr  = wave + 1 + dy;
      const int ccb = npix + 1 + dx;
      // B fragment: two 16B LDS loads (K-contiguous f16 layout)
      const _Float16* bp = smem + ((rr * 18 + ccb) << 5) + khalf;
      v8h blo = *(const v8h*)bp;
      v8h bhi = *(const v8h*)(bp + 16);
      v16h b = __builtin_shufflevector(blo, bhi, 0, 1, 2, 3, 4, 5, 6, 7,
                                       8, 9, 10, 11, 12, 13, 14, 15);
      // A fragments: pre-repacked WMMA-native layout, contiguous per lane
      const _Float16* wp0 = wpack + (((long)ot0 * T + t) * KC + kc) * 512 + lane * 16;
      v16h a0 = *(const v16h*)wp0;
      acc0 = __builtin_amdgcn_wmma_f32_16x16x32_f16(false, a0, false, b, (short)0, acc0,
                                                    false, false);
      if (has2) {
        v16h a1 = *(const v16h*)(wp0 + (long)T * KC * 512);
        acc1 = __builtin_amdgcn_wmma_f32_16x16x32_f16(false, a1, false, b, (short)0, acc1,
                                                      false, false);
      }
    }
  }

  // epilogue: BN fold (eval) / bias, ReLU; f32 NCHW store + f16 NHWC shadow
  float* outN = out + (long)n * OC * HWl;
  const int xg = x0 + npix;
  const long pixbase = (long)y * W + xg;
  {
    v8h hv = {};
#pragma unroll
    for (int r = 0; r < 8; ++r) {
      int oc = (ot0 << 4) + r + khalf;  // lanes>=16 hold M=r+8
      float valv = acc0[r];
      if (oc < OC) {
        if (bn_g) {
          float s = bn_g[oc] * rsqrtf(bn_v[oc] + 1e-5f);
          valv = valv * s + (bn_b[oc] - bn_m[oc] * s);
        } else if (bias) {
          valv += bias[oc];
        }
        if (relu) valv = fmaxf(valv, 0.f);
        outN[(long)oc * HWl + pixbase] = valv;
      } else {
        valv = 0.f;
      }
      hv[r] = (_Float16)valv;
    }
    if (out_h)  // OC % 32 == 0 for shadowed tensors: 8 consecutive channels
      *(v8h*)(out_h + ((long)n * HWl + pixbase) * OC + (ot0 << 4) + khalf) = hv;
  }
  if (has2) {
    v8h hv = {};
#pragma unroll
    for (int r = 0; r < 8; ++r) {
      int oc = ((ot0 + 1) << 4) + r + khalf;
      float valv = acc1[r];
      if (oc < OC) {
        if (bn_g) {
          float s = bn_g[oc] * rsqrtf(bn_v[oc] + 1e-5f);
          valv = valv * s + (bn_b[oc] - bn_m[oc] * s);
        } else if (bias) {
          valv += bias[oc];
        }
        if (relu) valv = fmaxf(valv, 0.f);
        outN[(long)oc * HWl + pixbase] = valv;
      } else {
        valv = 0.f;
      }
      hv[r] = (_Float16)valv;
    }
    if (out_h)
      *(v8h*)(out_h + ((long)n * HWl + pixbase) * OC + ((ot0 + 1) << 4) + khalf) = hv;
  }
}

// ========================== elementwise kernels =============================
// relu + 2x2 maxpool; optional f16 NHWC shadow (Cpad == C, C % 32 == 0)
__global__ void relu_pool_kernel(const float* __restrict__ in, float* __restrict__ out,
                                 _Float16* __restrict__ out_h, int N, int C, int H, int W) {
  int oh = H >> 1, ow = W >> 1;
  long total = (long)N * C * oh * ow;
  for (long i = (long)blockIdx.x * blockDim.x + threadIdx.x; i < total;
       i += (long)gridDim.x * blockDim.x) {
    int x = (int)(i % ow);
    long t = i / ow;
    int yy = (int)(t % oh); t /= oh;
    int c = (int)(t % C);
    int n = (int)(t / C);
    const float* p = in + (((long)n * C + c) * H + 2 * yy) * W + 2 * x;
    float a = fmaxf(p[0], 0.f), b = fmaxf(p[1], 0.f);
    float cc = fmaxf(p[W], 0.f), d = fmaxf(p[W + 1], 0.f);
    float v = fmaxf(fmaxf(a, b), fmaxf(cc, d));
    out[i] = v;
    if (out_h)
      out_h[((long)(n * oh + yy) * ow + x) * C + c] = (_Float16)v;
  }
}

// channel-concat copy; optional f16 NHWC shadow of the destination
__global__ void copy_chan_kernel(const float* __restrict__ src, float* __restrict__ dst,
                                 _Float16* __restrict__ dst_h,
                                 int N, int Cs, int Cd, int coff, long HW) {
  long total = (long)N * Cs * HW;
  for (long i = (long)blockIdx.x * blockDim.x + threadIdx.x; i < total;
       i += (long)gridDim.x * blockDim.x) {
    long hw = i % HW;
    long t = i / HW;
    int c = (int)(t % Cs);
    int n = (int)(t / Cs);
    float v = src[i];
    dst[((long)n * Cd + coff + c) * HW + hw] = v;
    if (dst_h)
      dst_h[((long)n * HW + hw) * Cd + coff + c] = (_Float16)v;
  }
}

// bilinear, align_corners=True; optional clamp to [-1,1]
__global__ void resize_ac_kernel(const float* __restrict__ in, float* __restrict__ out,
                                 int N, int C, int H, int W, int OH, int OW, int clip) {
  long total = (long)N * C * OH * OW;
  float sy = (OH > 1) ? (float)(H - 1) / (float)(OH - 1) : 0.f;
  float sx = (OW > 1) ? (float)(W - 1) / (float)(OW - 1) : 0.f;
  for (long i = (long)blockIdx.x * blockDim.x + threadIdx.x; i < total;
       i += (long)gridDim.x * blockDim.x) {
    int ox = (int)(i % OW);
    long t = i / OW;
    int oy = (int)(t % OH); t /= OH;
    int c = (int)(t % C);
    int n = (int)(t / C);
    float fy = oy * sy, fx = ox * sx;
    int y0 = (int)fy; if (y0 > H - 1) y0 = H - 1;
    int y1 = y0 + 1;  if (y1 > H - 1) y1 = H - 1;
    int x0 = (int)fx; if (x0 > W - 1) x0 = W - 1;
    int x1 = x0 + 1;  if (x1 > W - 1) x1 = W - 1;
    float wy = fy - (float)y0, wx = fx - (float)x0;
    const float* p = in + ((long)n * C + c) * H * W;
    float v00 = p[(long)y0 * W + x0], v01 = p[(long)y0 * W + x1];
    float v10 = p[(long)y1 * W + x0], v11 = p[(long)y1 * W + x1];
    float top = v00 + (v01 - v00) * wx;
    float bot = v10 + (v11 - v10) * wx;
    float v = top + (bot - top) * wy;
    if (clip) v = fminf(1.f, fmaxf(-1.f, v));
    out[i] = v;
  }
}

// bilinear, half-pixel centers (align_corners=False) for the final resizes
__global__ void resize_hp_kernel(const float* __restrict__ in, float* __restrict__ out,
                                 int N, int C, int H, int W, int OH, int OW) {
  long total = (long)N * C * OH * OW;
  float sy = (float)H / (float)OH;
  float sx = (float)W / (float)OW;
  for (long i = (long)blockIdx.x * blockDim.x + threadIdx.x; i < total;
       i += (long)gridDim.x * blockDim.x) {
    int ox = (int)(i % OW);
    long t = i / OW;
    int oy = (int)(t % OH); t /= OH;
    int c = (int)(t % C);
    int n = (int)(t / C);
    float fy = (oy + 0.5f) * sy - 0.5f;
    float fx = (ox + 0.5f) * sx - 0.5f;
    fy = fminf(fmaxf(fy, 0.f), (float)(H - 1));
    fx = fminf(fmaxf(fx, 0.f), (float)(W - 1));
    int y0 = (int)fy; if (y0 > H - 1) y0 = H - 1;
    int y1 = y0 + 1;  if (y1 > H - 1) y1 = H - 1;
    int x0 = (int)fx; if (x0 > W - 1) x0 = W - 1;
    int x1 = x0 + 1;  if (x1 > W - 1) x1 = W - 1;
    float wy = fy - (float)y0, wx = fx - (float)x0;
    const float* p = in + ((long)n * C + c) * H * W;
    float v00 = p[(long)y0 * W + x0], v01 = p[(long)y0 * W + x1];
    float v10 = p[(long)y1 * W + x0], v11 = p[(long)y1 * W + x1];
    float top = v00 + (v01 - v00) * wx;
    float bot = v10 + (v11 - v10) * wx;
    out[i] = top + (bot - top) * wy;
  }
}

// delta = softplus(dlin + lam*prior); Abar = exp(delta*(-exp(A[c])));
// Bbar = delta * (blin * (1 + alpha*prior))
__global__ void ssm_prep_kernel(const float* __restrict__ dlin, const float* __restrict__ blin,
                                const float* __restrict__ prior, const float* __restrict__ Avec,
                                const float* __restrict__ lam, const float* __restrict__ alpha,
                                float* __restrict__ Abar, float* __restrict__ Bbar,
                                int N, int Cx, int HW) {
  long total = (long)N * Cx * HW;
  float lamv = lam[0], alv = alpha[0];
  for (long i = (long)blockIdx.x * blockDim.x + threadIdx.x; i < total;
       i += (long)gridDim.x * blockDim.x) {
    long hw = i % HW;
    long t = i / HW;
    int c = (int)(t % Cx);
    int n = (int)(t / Cx);
    float pr = prior[(long)n * HW + hw];
    float d = dlin[i] + lamv * pr;
    d = (d > 20.f) ? d : log1pf(__expf(d));
    float A = -__expf(Avec[c]);
    Abar[i] = __expf(d * A);
    Bbar[i] = d * (blin[i] * (1.f + alv * pr));
  }
}

// recurrence along W: h_x = a_x*h_{x-1} + b_x*xp_x ; one thread per (n,c,row)
__global__ void scan_w_kernel(const float* __restrict__ xp, const float* __restrict__ a,
                              const float* __restrict__ b, float* __restrict__ hsum,
                              int N, int C, int H, int W) {
  int idx = blockIdx.x * blockDim.x + threadIdx.x;
  int total = N * C * H;
  if (idx >= total) return;
  long base = (long)idx * W;
  float h = 0.f;
  for (int x = 0; x < W; ++x) {
    long i = base + x;
    h = a[i] * h + b[i] * xp[i];
    hsum[i] = h;
  }
}

// recurrence along H (accumulates into hsum); one thread per (n,c,col).
// Writes the f16 NHWC shadow of the final sum (Cpad == C).
__global__ void scan_h_kernel(const float* __restrict__ xp, const float* __restrict__ a,
                              const float* __restrict__ b, float* __restrict__ hsum,
                              _Float16* __restrict__ hsum_h, int N, int C, int H, int W) {
  int idx = blockIdx.x * blockDim.x + threadIdx.x;
  int total = N * C * W;
  if (idx >= total) return;
  int x = idx % W;
  int nc = idx / W;
  int c = nc % C;
  int n = nc / C;
  long base = (long)nc * H * W + x;
  float h = 0.f;
  for (int y = 0; y < H; ++y) {
    long i = base + (long)y * W;
    h = a[i] * h + b[i] * xp[i];
    float v = hsum[i] + h;
    hsum[i] = v;
    if (hsum_h)
      hsum_h[((long)(n * H + y) * W + x) * C + c] = (_Float16)v;
  }
}

// out = x + gamma*o
__global__ void axpy_kernel(const float* __restrict__ x, const float* __restrict__ o,
                            const float* __restrict__ gamma, float* __restrict__ out, long n) {
  float g = gamma[0];
  for (long i = (long)blockIdx.x * blockDim.x + threadIdx.x; i < n;
       i += (long)gridDim.x * blockDim.x)
    out[i] = x[i] + g * o[i];
}

// ============================== host side ===================================
static inline int elw_blocks(long n) {
  long b = (n + 255) / 256;
  if (b > 32768) b = 32768;
  if (b < 1) b = 1;
  return (int)b;
}

extern "C" void kernel_launch(void* const* d_in, const int* in_sizes, int n_in,
                              void* d_out, int out_size, void* d_ws, size_t ws_size,
                              hipStream_t stream) {
  (void)in_sizes; (void)n_in; (void)out_size; (void)ws_size;
  const int N = 4;

  // ---------------- input unpack (setup_inputs dict order) ----------------
  int cur = 0;
  auto nxt = [&]() -> const float* { return (const float*)d_in[cur++]; };
  const float* imgA = nxt();
  const float* imgB = nxt();
  struct BN { const float *w, *g, *b, *m, *v; };
  auto rdbn = [&](BN& p) { p.w = nxt(); p.g = nxt(); p.b = nxt(); p.m = nxt(); p.v = nxt(); };
  BN vgg[13];
  for (int i = 0; i < 13; ++i) rdbn(vgg[i]);
  BN red1, red2, red3, red4, dc1;
  rdbn(red1); rdbn(red2); rdbn(red3); rdbn(red4); rdbn(dc1);
  const float* dc2_w = nxt(); const float* dc2_b = nxt();
  struct SSMP { const float *wi, *bi, *wo, *bo, *wd, *bd, *wb, *bb, *lam, *alpha, *A, *gamma; };
  auto rdssm = [&](SSMP& s) {
    s.wi = nxt(); s.bi = nxt(); s.wo = nxt(); s.bo = nxt();
    s.wd = nxt(); s.bd = nxt(); s.wb = nxt(); s.bb = nxt();
    s.lam = nxt(); s.alpha = nxt(); s.A = nxt(); s.gamma = nxt();
  };
  SSMP ssm3, ssm2, ssm1;
  rdssm(ssm3); rdssm(ssm2); rdssm(ssm1);
  BN dm4, dm3, dm2, fh1;
  rdbn(dm4); rdbn(dm3); rdbn(dm2); rdbn(fh1);
  const float* fh2_w = nxt(); const float* fh2_b = nxt();

  // ---------------- workspace bump allocator ----------------
  char* base = (char*)d_ws;
  size_t off = 0;
  auto alloc = [&](size_t bytes) -> void* {
    off = (off + 255) & ~(size_t)255;
    void* p = base + off;
    off += bytes;
    return p;
  };
  auto allocF = [&](size_t n) -> float* { return (float*)alloc(n * sizeof(float)); };
  auto allocH = [&](size_t n) -> _Float16* { return (_Float16*)alloc(n * sizeof(_Float16)); };
  auto markp = [&]() { return off; };
  auto popto = [&](size_t m) { off = m; };

  // ---------------- weight repack ----------------
  auto repack = [&](const float* w, int O, int I, int T) -> const _Float16* {
    int OT = (O + 15) / 16, KC = (I + 31) / 32;
    long total = (long)OT * T * KC * 512;
    _Float16* wp = (_Float16*)alloc((size_t)total * sizeof(_Float16));
    repack_w_kernel<<<elw_blocks(total), 256, 0, stream>>>(w, wp, O, I, T, total);
    return wp;
  };
  const int vggI[13] = {3, 64, 64, 128, 128, 256, 256, 256, 512, 512, 512, 512, 512};
  const int vggO[13] = {64, 64, 128, 128, 256, 256, 256, 512, 512, 512, 512, 512, 512};
  const _Float16* vggW[13];
  for (int i = 0; i < 13; ++i) vggW[i] = repack(vgg[i].w, vggO[i], vggI[i], 9);
  const _Float16* red1W = repack(red1.w, 128, 256, 9);
  const _Float16* red2W = repack(red2.w, 256, 512, 9);
  const _Float16* red3W = repack(red3.w, 512, 1024, 9);
  const _Float16* red4W = repack(red4.w, 512, 1024, 9);
  const _Float16* dc1W = repack(dc1.w, 64, 512, 9);
  const _Float16* dc2W = repack(dc2_w, 1, 64, 9);
  const _Float16* dm4W = repack(dm4.w, 512, 1024, 9);
  const _Float16* dm3W = repack(dm3.w, 256, 768, 9);
  const _Float16* dm2W = repack(dm2.w, 128, 384, 9);
  const _Float16* fh1W = repack(fh1.w, 64, 128, 9);
  const _Float16* fh2W = repack(fh2_w, 1, 64, 1);
  const _Float16* s3wi = repack(ssm3.wi, 256, 512, 1);
  const _Float16* s3wd = repack(ssm3.wd, 256, 256, 1);
  const _Float16* s3wb = repack(ssm3.wb, 256, 256, 1);
  const _Float16* s3wo = repack(ssm3.wo, 512, 256, 1);
  const _Float16* s2wi = repack(ssm2.wi, 128, 256, 1);
  const _Float16* s2wd = repack(ssm2.wd, 128, 128, 1);
  const _Float16* s2wb = repack(ssm2.wb, 128, 128, 1);
  const _Float16* s2wo = repack(ssm2.wo, 256, 128, 1);
  const _Float16* s1wi = repack(ssm1.wi, 64, 128, 1);
  const _Float16* s1wd = repack(ssm1.wd, 64, 64, 1);
  const _Float16* s1wb = repack(ssm1.wb, 64, 64, 1);
  const _Float16* s1wo = repack(ssm1.wo, 128, 64, 1);

  // ---------------- launchers ----------------
  auto conv = [&](const float* in, const _Float16* in_h, const _Float16* wp,
                  const float* g, const float* b, const float* m, const float* v,
                  const float* bias, float* out, _Float16* out_h,
                  int C, int H, int W, int OC, int Kk, int relu) {
    int OCT = (OC + 15) / 16;
    int OCPT = (OCT + 1) / 2;
    dim3 grid(W / 16, H / 8, N * OCPT);
    if (Kk == 3)
      conv_wmma_kernel<3><<<grid, 256, 0, stream>>>(in, in_h, wp, g, b, m, v, bias,
                                                    out, out_h, N, C, H, W, OC, relu);
    else
      conv_wmma_kernel<1><<<grid, 256, 0, stream>>>(in, in_h, wp, g, b, m, v, bias,
                                                    out, out_h, N, C, H, W, OC, relu);
  };
  auto convBN = [&](const float* in, const _Float16* in_h, const _Float16* wp, const BN& p,
                    float* out, _Float16* out_h, int C, int H, int W, int OC, int relu) {
    conv(in, in_h, wp, p.g, p.b, p.m, p.v, nullptr, out, out_h, C, H, W, OC, 3, relu);
  };
  auto conv1x1 = [&](const float* in, const _Float16* in_h, const _Float16* wp,
                     const float* bias, float* out, _Float16* out_h,
                     int C, int H, int W, int OC) {
    conv(in, in_h, wp, nullptr, nullptr, nullptr, nullptr, bias, out, out_h,
         C, H, W, OC, 1, 0);
  };
  auto relu_pool = [&](const float* in, float* out, _Float16* out_h, int C, int H, int W) {
    long total = (long)N * C * (H / 2) * (W / 2);
    relu_pool_kernel<<<elw_blocks(total), 256, 0, stream>>>(in, out, out_h, N, C, H, W);
  };
  auto cat2 = [&](const float* a, const float* b, float* dst, _Float16* dst_h,
                  int Ca, int Cb, long HW) {
    copy_chan_kernel<<<elw_blocks((long)N * Ca * HW), 256, 0, stream>>>(
        a, dst, dst_h, N, Ca, Ca + Cb, 0, HW);
    copy_chan_kernel<<<elw_blocks((long)N * Cb * HW), 256, 0, stream>>>(
        b, dst, dst_h, N, Cb, Ca + Cb, Ca, HW);
  };
  auto resize_ac = [&](const float* in, float* out, int C, int H, int W, int OH, int OW,
                       int clip) {
    long total = (long)N * C * OH * OW;
    resize_ac_kernel<<<elw_blocks(total), 256, 0, stream>>>(in, out, N, C, H, W, OH, OW, clip);
  };

  // ---------------- persistent activations (+ f16 NHWC shadows) ----------------
  float* l1A = allocF((size_t)N * 128 * 128 * 128);
  float* l2A = allocF((size_t)N * 256 * 64 * 64);
  float* l3A = allocF((size_t)N * 512 * 32 * 32);
  float* l4A = allocF((size_t)N * 512 * 16 * 16);
  float* l1B = allocF((size_t)N * 128 * 128 * 128);
  float* l2B = allocF((size_t)N * 256 * 64 * 64);
  float* l3B = allocF((size_t)N * 512 * 32 * 32);
  float* l4B = allocF((size_t)N * 512 * 16 * 16);
  float* l1r = allocF((size_t)N * 128 * 128 * 128);
  float* l2r = allocF((size_t)N * 256 * 64 * 64);
  float* l3r = allocF((size_t)N * 512 * 32 * 32);
  float* l4r = allocF((size_t)N * 512 * 16 * 16);
  _Float16* l1rh = allocH((size_t)N * 128 * 128 * 128);
  _Float16* l2rh = allocH((size_t)N * 256 * 64 * 64);
  _Float16* l3rh = allocH((size_t)N * 512 * 32 * 32);
  _Float16* l4rh = allocH((size_t)N * 512 * 16 * 16);
  float* l1s = allocF((size_t)N * 128 * 128 * 128);
  float* l2s = allocF((size_t)N * 256 * 64 * 64);
  float* l3s = allocF((size_t)N * 512 * 32 * 32);
  float* f4  = allocF((size_t)N * 512 * 32 * 32);
  float* f3  = allocF((size_t)N * 256 * 64 * 64);
  float* f2  = allocF((size_t)N * 128 * 128 * 128);
  _Float16* f2h = allocH((size_t)N * 128 * 128 * 128);
  float* coarse = allocF((size_t)N * 16 * 16);
  float* fineb  = allocF((size_t)N * 128 * 128);

  // ---------------- backbone (both streams) ----------------
  {
    size_t m0 = markp();
    float* t0 = allocF((size_t)N * 64 * 256 * 256);
    float* t1 = allocF((size_t)N * 64 * 256 * 256);
    _Float16* t0h = allocH((size_t)N * 64 * 256 * 256);
    _Float16* t1h = allocH((size_t)N * 64 * 256 * 256);
    auto run_bb = [&](const float* img, float* L1, float* L2, float* L3, float* L4) {
      convBN(img, nullptr, vggW[0], vgg[0], t0, t0h, 3, 256, 256, 64, 1);
      convBN(t0, t0h, vggW[1], vgg[1], t1, nullptr, 64, 256, 256, 64, 0);
      relu_pool(t1, t0, t0h, 64, 256, 256);
      convBN(t0, t0h, vggW[2], vgg[2], t1, t1h, 64, 128, 128, 128, 1);
      convBN(t1, t1h, vggW[3], vgg[3], L1, nullptr, 128, 128, 128, 128, 0);
      relu_pool(L1, t0, t0h, 128, 128, 128);
      convBN(t0, t0h, vggW[4], vgg[4], t1, t1h, 128, 64, 64, 256, 1);
      convBN(t1, t1h, vggW[5], vgg[5], t0, t0h, 256, 64, 64, 256, 1);
      convBN(t0, t0h, vggW[6], vgg[6], L2, nullptr, 256, 64, 64, 256, 0);
      relu_pool(L2, t0, t0h, 256, 64, 64);
      convBN(t0, t0h, vggW[7], vgg[7], t1, t1h, 256, 32, 32, 512, 1);
      convBN(t1, t1h, vggW[8], vgg[8], t0, t0h, 512, 32, 32, 512, 1);
      convBN(t0, t0h, vggW[9], vgg[9], L3, nullptr, 512, 32, 32, 512, 0);
      relu_pool(L3, t0, t0h, 512, 32, 32);
      convBN(t0, t0h, vggW[10], vgg[10], t1, t1h, 512, 16, 16, 512, 1);
      convBN(t1, t1h, vggW[11], vgg[11], t0, t0h, 512, 16, 16, 512, 1);
      convBN(t0, t0h, vggW[12], vgg[12], L4, nullptr, 512, 16, 16, 512, 0);
    };
    run_bb(imgA, l1A, l2A, l3A, l4A);
    run_bb(imgB, l1B, l2B, l3B, l4B);
    popto(m0);
  }

  // ---------------- concat + reduction convs ----------------
  auto reduce = [&](const float* a, const float* b, int C, int H, int W,
                    const _Float16* wR, const BN& pR, int OCr,
                    float* outr, _Float16* outrh) {
    size_t m = markp();
    float* cat = allocF((size_t)N * 2 * C * H * W);
    _Float16* cath = allocH((size_t)N * 2 * C * H * W);
    cat2(a, b, cat, cath, C, C, (long)H * W);
    convBN(cat, cath, wR, pR, outr, outrh, 2 * C, H, W, OCr, 1);
    popto(m);
  };
  reduce(l1A, l1B, 128, 128, 128, red1W, red1, 128, l1r, l1rh);
  reduce(l2A, l2B, 256, 64, 64, red2W, red2, 256, l2r, l2rh);
  reduce(l3A, l3B, 512, 32, 32, red3W, red3, 512, l3r, l3rh);
  reduce(l4A, l4B, 512, 16, 16, red4W, red4, 512, l4r, l4rh);

  // ---------------- coarse head ----------------
  {
    size_t m = markp();
    float* t = allocF((size_t)N * 64 * 16 * 16);
    _Float16* th = allocH((size_t)N * 64 * 16 * 16);
    convBN(l4r, l4rh, dc1W, dc1, t, th, 512, 16, 16, 64, 1);
    conv(t, th, dc2W, nullptr, nullptr, nullptr, nullptr, dc2_b, coarse, nullptr,
         64, 16, 16, 1, 3, 0);
    popto(m);
  }

  // ---------------- output layout ----------------
  float* dout = (float*)d_out;
  float* out_coarse = dout;                      // 4*1*256*256
  float* out_fine = dout + (size_t)N * 65536;    // 4*1*256*256
  float* out_g1 = out_fine + (size_t)N * 65536;  // 4*1*128*128
  float* out_g2 = out_g1 + (size_t)N * 16384;    // 4*1*64*64
  float* out_g3 = out_g2 + (size_t)N * 4096;     // 4*1*32*32

  // ---------------- SSM module ----------------
  auto run_ssm = [&](const float* x, const _Float16* xh, int Cin, int hid, int Hs, int Ws,
                     const SSMP& sp, const _Float16* wiP, const _Float16* wdP,
                     const _Float16* wbP, const _Float16* woP,
                     float* ls, float* gate_out) {
    size_t m = markp();
    long HW = (long)Hs * Ws;
    float* prior = allocF((size_t)N * HW);
    resize_ac(coarse, prior, 1, 16, 16, Hs, Ws, 1);
    (void)hipMemcpyAsync(gate_out, prior, (size_t)N * HW * sizeof(float),
                         hipMemcpyDeviceToDevice, stream);
    float* xp = allocF((size_t)N * hid * HW);
    _Float16* xph = allocH((size_t)N * hid * HW);
    conv1x1(x, xh, wiP, sp.bi, xp, xph, Cin, Hs, Ws, hid);
    float* dl = allocF((size_t)N * hid * HW);
    conv1x1(xp, xph, wdP, sp.bd, dl, nullptr, hid, Hs, Ws, hid);
    float* bl = allocF((size_t)N * hid * HW);
    conv1x1(xp, xph, wbP, sp.bb, bl, nullptr, hid, Hs, Ws, hid);
    float* Ab = allocF((size_t)N * hid * HW);
    float* Bb = allocF((size_t)N * hid * HW);
    ssm_prep_kernel<<<elw_blocks((long)N * hid * HW), 256, 0, stream>>>(
        dl, bl, prior, sp.A, sp.lam, sp.alpha, Ab, Bb, N, hid, (int)HW);
    float* hs = allocF((size_t)N * hid * HW);
    _Float16* hsh = allocH((size_t)N * hid * HW);
    {
      int tw = N * hid * Hs;
      scan_w_kernel<<<(tw + 255) / 256, 256, 0, stream>>>(xp, Ab, Bb, hs, N, hid, Hs, Ws);
      int th = N * hid * Ws;
      scan_h_kernel<<<(th + 255) / 256, 256, 0, stream>>>(xp, Ab, Bb, hs, hsh, N, hid, Hs, Ws);
    }
    float* oc_ = allocF((size_t)N * Cin * HW);
    conv1x1(hs, hsh, woP, sp.bo, oc_, nullptr, hid, Hs, Ws, Cin);
    axpy_kernel<<<elw_blocks((long)N * Cin * HW), 256, 0, stream>>>(
        x, oc_, sp.gamma, ls, (long)N * Cin * HW);
    popto(m);
  };

  // ---------------- decoder ----------------
  run_ssm(l3r, l3rh, 512, 256, 32, 32, ssm3, s3wi, s3wd, s3wb, s3wo, l3s, out_g3);
  {
    size_t m = markp();
    float* up4 = allocF((size_t)N * 512 * 32 * 32);
    resize_ac(l4r, up4, 512, 16, 16, 32, 32, 0);
    float* cat = allocF((size_t)N * 1024 * 32 * 32);
    _Float16* cath = allocH((size_t)N * 1024 * 32 * 32);
    cat2(up4, l3s, cat, cath, 512, 512, 32L * 32);
    convBN(cat, cath, dm4W, dm4, f4, nullptr, 1024, 32, 32, 512, 1);
    popto(m);
  }
  run_ssm(l2r, l2rh, 256, 128, 64, 64, ssm2, s2wi, s2wd, s2wb, s2wo, l2s, out_g2);
  {
    size_t m = markp();
    float* upf = allocF((size_t)N * 512 * 64 * 64);
    resize_ac(f4, upf, 512, 32, 32, 64, 64, 0);
    float* cat = allocF((size_t)N * 768 * 64 * 64);
    _Float16* cath = allocH((size_t)N * 768 * 64 * 64);
    cat2(upf, l2s, cat, cath, 512, 256, 64L * 64);
    convBN(cat, cath, dm3W, dm3, f3, nullptr, 768, 64, 64, 256, 1);
    popto(m);
  }
  run_ssm(l1r, l1rh, 128, 64, 128, 128, ssm1, s1wi, s1wd, s1wb, s1wo, l1s, out_g1);
  {
    size_t m = markp();
    float* upf = allocF((size_t)N * 256 * 128 * 128);
    resize_ac(f3, upf, 256, 64, 64, 128, 128, 0);
    float* cat = allocF((size_t)N * 384 * 128 * 128);
    _Float16* cath = allocH((size_t)N * 384 * 128 * 128);
    cat2(upf, l1s, cat, cath, 256, 128, 128L * 128);
    convBN(cat, cath, dm2W, dm2, f2, f2h, 384, 128, 128, 128, 1);
    popto(m);
  }
  // fine head
  {
    size_t m = markp();
    float* t = allocF((size_t)N * 64 * 128 * 128);
    _Float16* th = allocH((size_t)N * 64 * 128 * 128);
    convBN(f2, f2h, fh1W, fh1, t, th, 128, 128, 128, 64, 1);
    conv1x1(t, th, fh2W, fh2_b, fineb, nullptr, 64, 128, 128, 1);
    popto(m);
  }
  // final resizes (align_corners=False) into d_out
  {
    long tc = (long)N * 256 * 256;
    resize_hp_kernel<<<elw_blocks(tc), 256, 0, stream>>>(coarse, out_coarse, N, 1, 16, 16,
                                                         256, 256);
    resize_hp_kernel<<<elw_blocks(tc), 256, 0, stream>>>(fineb, out_fine, N, 1, 128, 128,
                                                         256, 256);
  }
}
